// MultiHeadAttention_90168543412663
// MI455X (gfx1250) — matrix operand order repaired and checked by reading, not compiled, not run on hardware
//
#include <hip/hip_runtime.h>
#include <hip/hip_bf16.h>

#define B_   2
#define S_   2048
#define HID_ 2048
#define NH_  32
#define NKV_ 8
#define HD_  64
#define NREP_ 4
#define SCALE_ 0.125f

typedef __attribute__((ext_vector_type(16))) __bf16 v16bf;
typedef __attribute__((ext_vector_type(8)))  float  v8f;
typedef __attribute__((ext_vector_type(4)))  float  v4f;

union Frag { v16bf v; v4f f4[2]; };

#define WMMA_BF16(a, b, c) \
  __builtin_amdgcn_wmma_f32_16x16x32_bf16(false, (a), false, (b), (short)0, (c), false, false)

// A fragment: 16x32 bf16, row m = lane%16, K chunks [kb+half*8, +8) and [kb+16+half*8, +8)
__device__ inline v16bf load_a_frag(const __bf16* A, int lda, int row0, int kb, int lane) {
  int m = lane & 15, hl = lane >> 4;
  const __bf16* p = A + (size_t)(row0 + m) * (size_t)lda + kb + hl * 8;
  Frag fr;
  fr.f4[0] = *(const v4f*)p;
  fr.f4[1] = *(const v4f*)(p + 16);
  return fr.v;
}

// B fragment: 32x16 (KxN) bf16 built from row-major source Bm[N][K] (i.e. B = Bm^T tile):
// col n = lane%16, K chunks [kb+half*8, +8) and [kb+16+half*8, +8)
__device__ inline v16bf load_b_frag(const __bf16* Bm, int ldb, int n0, int kb, int lane) {
  int n = lane & 15, hl = lane >> 4;
  const __bf16* p = Bm + (size_t)(n0 + n) * (size_t)ldb + kb + hl * 8;
  Frag fr;
  fr.f4[0] = *(const v4f*)p;
  fr.f4[1] = *(const v4f*)(p + 16);
  return fr.v;
}

__global__ void cvt_f32_bf16_kernel(const float* __restrict__ src, __bf16* __restrict__ dst, long n) {
  long i = (long)blockIdx.x * 256 + threadIdx.x;
  if (i < n) dst[i] = (__bf16)src[i];
}

// C[M,N] = A[M,K] * Bm[N,K]^T   (bf16 in, fp32 out). One wave per 64x32 tile:
// 4 A-frags x 2 B-frags -> 8 WMMAs per k=32 step, 8 named accumulators (no spills).
__global__ void gemm_bf16_nt_kernel(const __bf16* __restrict__ A, const __bf16* __restrict__ Bm,
                                    float* __restrict__ C, int M, int N, int K) {
  int wave = threadIdx.x >> 5;
  int lane = threadIdx.x & 31;
  int tiles_n = N >> 5;                     // N/32
  int tile = blockIdx.x * 8 + wave;
  int tm = tile / tiles_n;
  int tn = tile - tm * tiles_n;
  if (tm >= (M >> 6)) return;               // M/64
  int row0 = tm * 64, col0 = tn * 32;

  v8f c00 = {}; v8f c01 = {};
  v8f c10 = {}; v8f c11 = {};
  v8f c20 = {}; v8f c21 = {};
  v8f c30 = {}; v8f c31 = {};

  for (int kb = 0; kb < K; kb += 32) {
    if (kb + 64 < K) {
      // warm caches for the k-block after next (global_prefetch_b8)
      __builtin_prefetch(A + (size_t)(row0 + (lane & 15)) * K + kb + 64, 0, 1);
      __builtin_prefetch(Bm + (size_t)(col0 + (lane & 15)) * K + kb + 64, 0, 1);
    }
    v16bf a0 = load_a_frag(A, K, row0,      kb, lane);
    v16bf a1 = load_a_frag(A, K, row0 + 16, kb, lane);
    v16bf a2 = load_a_frag(A, K, row0 + 32, kb, lane);
    v16bf a3 = load_a_frag(A, K, row0 + 48, kb, lane);
    v16bf b0 = load_b_frag(Bm, K, col0,      kb, lane);
    v16bf b1 = load_b_frag(Bm, K, col0 + 16, kb, lane);
    c00 = WMMA_BF16(a0, b0, c00);
    c01 = WMMA_BF16(a0, b1, c01);
    c10 = WMMA_BF16(a1, b0, c10);
    c11 = WMMA_BF16(a1, b1, c11);
    c20 = WMMA_BF16(a2, b0, c20);
    c21 = WMMA_BF16(a2, b1, c21);
    c30 = WMMA_BF16(a3, b0, c30);
    c31 = WMMA_BF16(a3, b1, c31);
  }

  int hl = lane >> 4, n = lane & 15;
  for (int r = 0; r < 8; ++r) {
    int m0 = r + hl * 8;
    C[(size_t)(row0 + m0) * N + col0 + n]           = c00[r];
    C[(size_t)(row0 + m0) * N + col0 + 16 + n]      = c01[r];
    C[(size_t)(row0 + 16 + m0) * N + col0 + n]      = c10[r];
    C[(size_t)(row0 + 16 + m0) * N + col0 + 16 + n] = c11[r];
    C[(size_t)(row0 + 32 + m0) * N + col0 + n]      = c20[r];
    C[(size_t)(row0 + 32 + m0) * N + col0 + 16 + n] = c21[r];
    C[(size_t)(row0 + 48 + m0) * N + col0 + n]      = c30[r];
    C[(size_t)(row0 + 48 + m0) * N + col0 + 16 + n] = c31[r];
  }
}

// proj: [B,S,nheads,HD] fp32 -> dst: [B,nheads,S,HD] bf16, with RoPE (half-rotate).
__global__ void rope_pack_kernel(const float* __restrict__ proj,
                                 const float* __restrict__ cosv,
                                 const float* __restrict__ sinv,
                                 __bf16* __restrict__ dst, int nheads) {
  long i = (long)blockIdx.x * 256 + threadIdx.x;
  long total = (long)B_ * S_ * nheads * 32;
  if (i >= total) return;
  int d  = (int)(i & 31);
  long t = i >> 5;
  int h  = (int)(t % nheads);
  long bs = t / nheads;                 // b*S + s
  const float* row = proj + (bs * nheads + h) * HD_;
  float x1 = row[d], x2 = row[d + 32];
  float c1 = cosv[bs * HD_ + d],      s1 = sinv[bs * HD_ + d];
  float c2 = cosv[bs * HD_ + d + 32], s2 = sinv[bs * HD_ + d + 32];
  float o1 = x1 * c1 - x2 * s1;
  float o2 = x2 * c2 + x1 * s2;
  long b = bs / S_, s = bs % S_;
  __bf16* drow = dst + (((b * nheads + h) * S_) + s) * HD_;
  drow[d]      = (__bf16)o1;
  drow[d + 32] = (__bf16)o2;
}

// proj: [B,S,NKV,HD] fp32 -> dst: [B,NKV,HD,S] bf16 (transposed, no rope).
__global__ void pack_vt_kernel(const float* __restrict__ proj, __bf16* __restrict__ dst) {
  long i = (long)blockIdx.x * 256 + threadIdx.x;
  long total = (long)B_ * S_ * NKV_ * HD_;
  if (i >= total) return;
  int d  = (int)(i & (HD_ - 1));
  long t = i >> 6;
  int h  = (int)(t % NKV_);
  long bs = t / NKV_;
  long b = bs / S_, s = bs % S_;
  dst[(((b * NKV_ + h) * HD_) + d) * S_ + s] = (__bf16)proj[(bs * NKV_ + h) * HD_ + d];
}

// Flash attention: 1 wave per 16 query rows. Q/K: [B,H,S,64] bf16, Vt: [B,NKV,64,S] bf16.
// Out: [B,S,NH*HD] bf16.
__global__ void flash_attn_kernel(const __bf16* __restrict__ Qbf,
                                  const __bf16* __restrict__ Kbf,
                                  const __bf16* __restrict__ Vt,
                                  __bf16* __restrict__ Obf) {
  __shared__ float  sc[16][32];
  __shared__ __bf16 pb[16][32];
  __shared__ float  mrow[16], lrow[16], arow[16];

  int lane = threadIdx.x;
  int q0 = blockIdx.x * 16;
  int h = blockIdx.y, b = blockIdx.z;
  int hkv = h / NREP_;

  const __bf16* Q  = Qbf + ((size_t)(b * NH_ + h) * S_) * HD_;
  const __bf16* Kp = Kbf + ((size_t)(b * NKV_ + hkv) * S_) * HD_;
  const __bf16* Vp = Vt  + ((size_t)(b * NKV_ + hkv) * HD_) * S_;

  v16bf aq0 = load_a_frag(Q, HD_, q0, 0, lane);
  v16bf aq1 = load_a_frag(Q, HD_, q0, 32, lane);

  v8f acc0 = {}; v8f acc1 = {}; v8f acc2 = {}; v8f acc3 = {};
  if (lane < 16) { mrow[lane] = -INFINITY; lrow[lane] = 0.f; }
  __syncthreads();

  int hl = lane >> 4;
  int nn = lane & 15;
  int kend = q0 + 16;                       // causal: keys < q0+16 can be unmasked
  for (int kb = 0; kb < kend; kb += 32) {
    // --- scores: two 16x16 tiles over this 32-key block ---
    for (int t = 0; t < 2; ++t) {
      v16bf bk0 = load_b_frag(Kp, HD_, kb + t * 16, 0, lane);
      v16bf bk1 = load_b_frag(Kp, HD_, kb + t * 16, 32, lane);
      v8f s = {};
      s = WMMA_BF16(aq0, bk0, s);
      s = WMMA_BF16(aq1, bk1, s);
      int key = kb + t * 16 + nn;
      for (int r = 0; r < 8; ++r) {
        int qrow = q0 + r + hl * 8;
        sc[r + hl * 8][t * 16 + nn] = s[r] * SCALE_ + (key <= qrow ? 0.f : -1e9f);
      }
    }
    __syncthreads();

    // --- online softmax over rows (lanes 0..15 own one row each) ---
    if (lane < 16) {
      float mold = mrow[lane], lold = lrow[lane];
      float mx = mold;
      for (int j = 0; j < 32; ++j) mx = fmaxf(mx, sc[lane][j]);
      float alpha = __expf(mold - mx);
      float sum = 0.f;
      for (int j = 0; j < 32; ++j) {
        float p = __expf(sc[lane][j] - mx);
        pb[lane][j] = (__bf16)p;
        sum += p;
      }
      mrow[lane] = mx;
      lrow[lane] = alpha * lold + sum;
      arow[lane] = alpha;
    }
    __syncthreads();

    // --- rescale accumulators by alpha(row) ---
    for (int r = 0; r < 8; ++r) {
      float al = arow[r + hl * 8];
      acc0[r] *= al; acc1[r] *= al; acc2[r] *= al; acc3[r] *= al;
    }

    // --- P (16x32, A-layout from LDS) @ V (32x64 via 4 B-frags from transposed V) ---
    Frag fr;
    fr.f4[0] = *(const v4f*)&pb[lane & 15][hl * 8];
    fr.f4[1] = *(const v4f*)&pb[lane & 15][16 + hl * 8];
    v16bf ap = fr.v;
    v16bf bv0 = load_b_frag(Vp, S_, 0,  kb, lane);
    v16bf bv1 = load_b_frag(Vp, S_, 16, kb, lane);
    v16bf bv2 = load_b_frag(Vp, S_, 32, kb, lane);
    v16bf bv3 = load_b_frag(Vp, S_, 48, kb, lane);
    acc0 = WMMA_BF16(ap, bv0, acc0);
    acc1 = WMMA_BF16(ap, bv1, acc1);
    acc2 = WMMA_BF16(ap, bv2, acc2);
    acc3 = WMMA_BF16(ap, bv3, acc3);
    __syncthreads();
  }

  // --- epilogue: out = acc / l ---
  for (int r = 0; r < 8; ++r) {
    int m = r + hl * 8;
    float linv = 1.0f / lrow[m];
    size_t orow = ((size_t)b * S_ + q0 + m) * (NH_ * HD_) + h * HD_;
    Obf[orow + 0  + nn] = (__bf16)(acc0[r] * linv);
    Obf[orow + 16 + nn] = (__bf16)(acc1[r] * linv);
    Obf[orow + 32 + nn] = (__bf16)(acc2[r] * linv);
    Obf[orow + 48 + nn] = (__bf16)(acc3[r] * linv);
  }
}

extern "C" void kernel_launch(void* const* d_in, const int* in_sizes, int n_in,
                              void* d_out, int out_size, void* d_ws, size_t ws_size,
                              hipStream_t stream) {
  (void)in_sizes; (void)n_in; (void)out_size; (void)ws_size;
  const float* hs   = (const float*)d_in[0];
  const float* cosv = (const float*)d_in[1];
  const float* sinv = (const float*)d_in[2];
  /* d_in[3] = attention_mask: causal -1e9 mask, applied analytically in-kernel */
  const float* Wq = (const float*)d_in[4];
  const float* Wk = (const float*)d_in[5];
  const float* Wv = (const float*)d_in[6];
  const float* Wo = (const float*)d_in[7];
  float* out = (float*)d_out;

  char* ws = (char*)d_ws;
  size_t off = 0;
  auto alloc = [&](size_t bytes) -> void* {
    void* p = ws + off;
    off += (bytes + 255) & ~(size_t)255;
    return p;
  };

  const size_t MS = (size_t)B_ * S_;  // 4096 rows
  __bf16* Xbf = (__bf16*)alloc(MS * HID_ * 2);
  __bf16* Wqb = (__bf16*)alloc((size_t)NH_ * HD_ * HID_ * 2);
  __bf16* Wkb = (__bf16*)alloc((size_t)NKV_ * HD_ * HID_ * 2);
  __bf16* Wvb = (__bf16*)alloc((size_t)NKV_ * HD_ * HID_ * 2);
  __bf16* Wob = (__bf16*)alloc((size_t)HID_ * NH_ * HD_ * 2);
  float*  Qp  = (float*)alloc(MS * NH_ * HD_ * 4);
  float*  Kp  = (float*)alloc(MS * NKV_ * HD_ * 4);
  float*  Vp  = (float*)alloc(MS * NKV_ * HD_ * 4);
  __bf16* Qbf = (__bf16*)alloc(MS * NH_ * HD_ * 2);
  __bf16* Kbf = (__bf16*)alloc(MS * NKV_ * HD_ * 2);
  __bf16* Vtb = (__bf16*)alloc(MS * NKV_ * HD_ * 2);
  __bf16* Obf = (__bf16*)alloc(MS * NH_ * HD_ * 2);

  auto cvt = [&](const float* s, __bf16* d, size_t n) {
    cvt_f32_bf16_kernel<<<(unsigned)((n + 255) / 256), 256, 0, stream>>>(s, d, (long)n);
  };
  cvt(hs, Xbf, MS * HID_);
  cvt(Wq, Wqb, (size_t)NH_ * HD_ * HID_);
  cvt(Wk, Wkb, (size_t)NKV_ * HD_ * HID_);
  cvt(Wv, Wvb, (size_t)NKV_ * HD_ * HID_);
  cvt(Wo, Wob, (size_t)HID_ * NH_ * HD_);

  auto gemm = [&](const __bf16* A, const __bf16* Bm, float* C, int M, int N, int K) {
    int tiles = (M / 64) * (N / 32);
    gemm_bf16_nt_kernel<<<tiles / 8, 256, 0, stream>>>(A, Bm, C, M, N, K);
  };
  gemm(Xbf, Wqb, Qp, (int)MS, NH_ * HD_, HID_);   // 4096 x 2048 x 2048
  gemm(Xbf, Wkb, Kp, (int)MS, NKV_ * HD_, HID_);  // 4096 x  512 x 2048
  gemm(Xbf, Wvb, Vp, (int)MS, NKV_ * HD_, HID_);

  rope_pack_kernel<<<(B_ * S_ * NH_ * 32) / 256, 256, 0, stream>>>(Qp, cosv, sinv, Qbf, NH_);
  rope_pack_kernel<<<(B_ * S_ * NKV_ * 32) / 256, 256, 0, stream>>>(Kp, cosv, sinv, Kbf, NKV_);
  pack_vt_kernel<<<(B_ * S_ * NKV_ * HD_) / 256, 256, 0, stream>>>(Vp, Vtb);

  flash_attn_kernel<<<dim3(S_ / 16, NH_, B_), 32, 0, stream>>>(Qbf, Kbf, Vtb, Obf);

  gemm(Obf, Wob, out, (int)MS, HID_, NH_ * HD_);  // final fp32 output
}
